// TF_MultiHeadAttention_31765578121488
// MI455X (gfx1250) — matrix operand order repaired and checked by reading, not compiled, not run on hardware
//
#include <hip/hip_runtime.h>
#include <hip/hip_bf16.h>
#include <cmath>

typedef __attribute__((ext_vector_type(16))) _Float16 v16h;
typedef __attribute__((ext_vector_type(8)))  _Float16 v8h;
typedef __attribute__((ext_vector_type(8)))  float    v8f;

#define BSZ 4
#define SLN 2048
#define DM  1024
#define NH  16
#define HD  64
#define BHN (BSZ*NH)

__device__ __forceinline__ v8f wmma_f16(v16h a, v16h b, v8f c) {
  // D = A(16x32 f16) * B(32x16 f16) + C(16x16 f32)
  return __builtin_amdgcn_wmma_f32_16x16x32_f16(
      /*neg_a=*/false, a, /*neg_b=*/false, b,
      /*c_mod=*/(short)0, c, /*reuse_a=*/false, /*reuse_b=*/false);
}

// Assemble a 16-half fragment from two contiguous 8-half runs (16B aligned).
__device__ __forceinline__ v16h make_frag(const _Float16* p0, const _Float16* p1) {
  v8h lo = *(const v8h*)p0;
  v8h hi = *(const v8h*)p1;
  v16h r;
#pragma unroll
  for (int i = 0; i < 8; ++i) { r[i] = lo[i]; r[i + 8] = hi[i]; }
  return r;
}

__global__ void f32_to_f16_kernel(const float* __restrict__ in,
                                  _Float16* __restrict__ out, int n) {
  int i = blockIdx.x * blockDim.x + threadIdx.x;
  if (i < n) out[i] = (_Float16)in[i];
}

// C[MxN] = A[MxK] * B[KxN] + bias.  A is f32 or f16; output either
// head-split f16 [B,H,SL,HD] (projections) or row-major f32 (final).
// BM=128, BN=64, BK=32; 4 waves, each owning a 32x64 strip (8 accumulators).
template <bool A_F32, bool OUT_HEADS>
__global__ __launch_bounds__(128) void gemm_wmma(
    const void* __restrict__ Ain, const _Float16* __restrict__ Bw,
    const float* __restrict__ bias, void* __restrict__ Cout,
    int M, int N, int K) {
  // 40-half row stride: 80B = 16B aligned, and r*20 mod 64 distinct for r=0..15
  // -> conflict-free 16-lane b128 fragment reads.
  __shared__ _Float16 As[128][40];
  __shared__ _Float16 Bt[64][40];    // B tile transposed: [n][k]

  const int tid = threadIdx.x;
  const int wave = tid >> 5, lane = tid & 31;
  const int lane15 = lane & 15, hi16 = lane >> 4;
  const int rowBase = blockIdx.y * 128;
  const int colBase = blockIdx.x * 64;

  v8f acc[2][4] = {};

  for (int k0 = 0; k0 < K; k0 += 32) {
    // --- load A tile: thread t loads its whole row (32 elems, vectorized) ---
    {
      int gr = rowBase + tid;
      if (A_F32) {
        const float* A = (const float*)Ain;
        const float4* src = (const float4*)(A + (size_t)gr * K + k0);
#pragma unroll
        for (int c = 0; c < 8; ++c) {
          float4 f = src[c];
          As[tid][c * 4 + 0] = (_Float16)f.x;
          As[tid][c * 4 + 1] = (_Float16)f.y;
          As[tid][c * 4 + 2] = (_Float16)f.z;
          As[tid][c * 4 + 3] = (_Float16)f.w;
        }
      } else {
        const _Float16* A = (const _Float16*)Ain;
        const v8h* src = (const v8h*)(A + (size_t)gr * K + k0);
#pragma unroll
        for (int c = 0; c < 4; ++c) *(v8h*)&As[tid][c * 8] = src[c];
      }
    }
    // --- load B tile: vectorized global b128 loads, transposed LDS stores ---
    {
      int kk = tid >> 2;                 // 0..31
      int n0 = (tid & 3) * 16;           // 0,16,32,48
      const _Float16* bsrc = Bw + (size_t)(k0 + kk) * N + colBase + n0;
      v8h b0 = *(const v8h*)(bsrc);
      v8h b1 = *(const v8h*)(bsrc + 8);
#pragma unroll
      for (int i = 0; i < 8; ++i) {
        Bt[n0 + i][kk] = b0[i];
        Bt[n0 + 8 + i][kk] = b1[i];
      }
    }
    __syncthreads();

    v16h afrag[2];
#pragma unroll
    for (int i = 0; i < 2; ++i) {
      int r = wave * 32 + i * 16 + lane15;
      int kb = hi16 * 8;
      afrag[i] = make_frag(&As[r][kb], &As[r][16 + kb]);
    }
#pragma unroll
    for (int j = 0; j < 4; ++j) {
      int n = j * 16 + lane15;
      int kb = hi16 * 8;
      v16h bfrag = make_frag(&Bt[n][kb], &Bt[n][kb + 16]);
#pragma unroll
      for (int i = 0; i < 2; ++i) acc[i][j] = wmma_f16(afrag[i], bfrag, acc[i][j]);
    }
    __syncthreads();
  }

  // --- epilogue: D layout: VGPR t -> M = t + 8*(lane>=16), N = lane&15 ---
#pragma unroll
  for (int i = 0; i < 2; ++i) {
#pragma unroll
    for (int j = 0; j < 4; ++j) {
#pragma unroll
      for (int t = 0; t < 8; ++t) {
        int m = wave * 32 + i * 16 + t + hi16 * 8;
        int gr = rowBase + m;
        int gc = colBase + j * 16 + lane15;
        float val = acc[i][j][t] + bias[gc];
        if (OUT_HEADS) {
          int b = gr / SLN, s = gr % SLN;
          int hh = gc >> 6, d = gc & 63;
          ((_Float16*)Cout)[((size_t)((b * NH + hh) * SLN + s)) * HD + d] =
              (_Float16)val;
        } else {
          ((float*)Cout)[(size_t)gr * N + gc] = val;
        }
      }
    }
  }
}

// Flash attention: block = 4 waves x 16 queries each; 64-key K/V tiles.
__global__ __launch_bounds__(128) void flash_attn(
    const _Float16* __restrict__ qh, const _Float16* __restrict__ kh,
    const _Float16* __restrict__ vh, const int* __restrict__ mask,
    _Float16* __restrict__ concat) {
  // 88-half row stride (176B, 16B-aligned): r*44 mod 64 distinct for r=0..15
  // -> conflict-free 16-lane b128 fragment reads.
  __shared__ _Float16 Ks[64][88];        // K tile [key][d]
  __shared__ _Float16 Vt[64][88];        // V tile transposed [d][key]
  __shared__ _Float16 Ps[4][16][88];     // per-wave P staging [row][key]

  const int tid = threadIdx.x;
  const int wave = tid >> 5, lane = tid & 31;
  const int lane15 = lane & 15, hi16 = lane >> 4;
  const int bh = blockIdx.y, qb = blockIdx.x;
  const int b = bh >> 4, hh = bh & 15;
  const size_t base = (size_t)bh * SLN * HD;
  const int q0 = qb * 64 + wave * 16;

  // Q fragments straight from global (row-major [q][d])
  v16h qf[2];
#pragma unroll
  for (int c = 0; c < 2; ++c) {
    const _Float16* qrow = qh + base + (size_t)(q0 + lane15) * HD + c * 32;
    int kb = hi16 * 8;
    qf[c] = make_frag(qrow + kb, qrow + 16 + kb);
  }

  v8f o[4] = {};
  float mrow[8], lrow[8];
#pragma unroll
  for (int t = 0; t < 8; ++t) { mrow[t] = -1e30f; lrow[t] = 0.f; }

  for (int kt = 0; kt < SLN / 64; ++kt) {
    __syncthreads();
    // cooperative load: 64x64 K tile row-major + V tile transposed.
    // thread t handles key = t>>1, d in [ (t&1)*32, +32 ): 4 b128 loads each.
    {
      int key = tid >> 1;
      int d0 = (tid & 1) * 32;
      const _Float16* ksrc = kh + base + (size_t)(kt * 64 + key) * HD + d0;
#pragma unroll
      for (int i = 0; i < 4; ++i)
        *(v8h*)&Ks[key][d0 + i * 8] = *(const v8h*)(ksrc + i * 8);
      const _Float16* vsrc = vh + base + (size_t)(kt * 64 + key) * HD + d0;
#pragma unroll
      for (int i = 0; i < 4; ++i) {
        v8h vv = *(const v8h*)(vsrc + i * 8);
#pragma unroll
        for (int u = 0; u < 8; ++u) Vt[d0 + i * 8 + u][key] = vv[u];
      }
    }
    __syncthreads();

    // S(16x64) = Q(16x64) * K^T : 4 subtiles x 2 chained WMMAs over d.
    v8f s[4] = {};
#pragma unroll
    for (int st = 0; st < 4; ++st) {
#pragma unroll
      for (int c = 0; c < 2; ++c) {
        int kb = c * 32 + hi16 * 8;
        v16h bk = make_frag(&Ks[st * 16 + lane15][kb],
                            &Ks[st * 16 + lane15][kb + 16]);
        s[st] = wmma_f16(qf[c], bk, s[st]);
      }
    }

    // scale, mask, online softmax (one shuffle-reduction per row per 64 keys)
#pragma unroll
    for (int t = 0; t < 8; ++t) {
      int qg = q0 + t + hi16 * 8;
      float sv[4];
#pragma unroll
      for (int st = 0; st < 4; ++st) {
        int kg = kt * 64 + st * 16 + lane15;
        sv[st] = s[st][t] * 0.125f;
        if (mask[(size_t)qg * SLN + kg] == 0) sv[st] = -10000.f;
      }
      float mx = fmaxf(fmaxf(sv[0], sv[1]), fmaxf(sv[2], sv[3]));
#pragma unroll
      for (int off = 1; off < 16; off <<= 1)
        mx = fmaxf(mx, __shfl_xor(mx, off, 16));
      float mnew = fmaxf(mrow[t], mx);
      float corr = __expf(mrow[t] - mnew);
      float p[4], ps = 0.f;
#pragma unroll
      for (int st = 0; st < 4; ++st) { p[st] = __expf(sv[st] - mnew); ps += p[st]; }
#pragma unroll
      for (int off = 1; off < 16; off <<= 1) ps += __shfl_xor(ps, off, 16);
      lrow[t] = lrow[t] * corr + ps;
      mrow[t] = mnew;
#pragma unroll
      for (int j = 0; j < 4; ++j) o[j][t] *= corr;
      int r = t + hi16 * 8;
#pragma unroll
      for (int st = 0; st < 4; ++st)
        Ps[wave][r][st * 16 + lane15] = (_Float16)p[st];
    }
    __syncthreads();

    // O(16x64) += P(16x64) * V(64x64): two 32-key chunks.
#pragma unroll
    for (int c = 0; c < 2; ++c) {
      int kb = hi16 * 8;
      v16h pa = make_frag(&Ps[wave][lane15][c * 32 + kb],
                          &Ps[wave][lane15][c * 32 + 16 + kb]);
#pragma unroll
      for (int j = 0; j < 4; ++j) {
        int kb2 = c * 32 + hi16 * 8;
        v16h bv = make_frag(&Vt[j * 16 + lane15][kb2],
                            &Vt[j * 16 + lane15][kb2 + 16]);
        o[j] = wmma_f16(pa, bv, o[j]);
      }
    }
  }

  // normalize and write concat [b*SL+s][h*64+d] as f16
#pragma unroll
  for (int j = 0; j < 4; ++j) {
#pragma unroll
    for (int t = 0; t < 8; ++t) {
      int s = q0 + t + hi16 * 8;
      int d = j * 16 + lane15;
      float val = o[j][t] / lrow[t];
      concat[((size_t)(b * SLN + s)) * (NH * HD) + hh * HD + d] = (_Float16)val;
    }
  }
}

extern "C" void kernel_launch(void* const* d_in, const int* in_sizes, int n_in,
                              void* d_out, int out_size, void* d_ws,
                              size_t ws_size, hipStream_t stream) {
  (void)in_sizes; (void)n_in; (void)out_size; (void)ws_size;
  const float* q  = (const float*)d_in[0];
  const float* k  = (const float*)d_in[1];
  const float* v  = (const float*)d_in[2];
  const int* mask = (const int*)d_in[3];
  const float* Wq = (const float*)d_in[4];
  const float* bq = (const float*)d_in[5];
  const float* Wk = (const float*)d_in[6];
  const float* bk = (const float*)d_in[7];
  const float* Wv = (const float*)d_in[8];
  const float* bv = (const float*)d_in[9];
  const float* Wo = (const float*)d_in[10];
  const float* bo = (const float*)d_in[11];
  float* out = (float*)d_out;

  char* ws = (char*)d_ws;
  size_t off = 0;
  auto alloc = [&](size_t bytes) -> char* {
    char* p = ws + off;
    off += (bytes + 255) & ~(size_t)255;
    return p;
  };
  _Float16* Wq_h = (_Float16*)alloc((size_t)DM * DM * 2);
  _Float16* Wk_h = (_Float16*)alloc((size_t)DM * DM * 2);
  _Float16* Wv_h = (_Float16*)alloc((size_t)DM * DM * 2);
  _Float16* Wo_h = (_Float16*)alloc((size_t)DM * DM * 2);
  _Float16* qh   = (_Float16*)alloc((size_t)BHN * SLN * HD * 2);
  _Float16* khb  = (_Float16*)alloc((size_t)BHN * SLN * HD * 2);
  _Float16* vhb  = (_Float16*)alloc((size_t)BHN * SLN * HD * 2);
  _Float16* cat  = (_Float16*)alloc((size_t)BSZ * SLN * DM * 2);

  const int nW = DM * DM;
  f32_to_f16_kernel<<<(nW + 255) / 256, 256, 0, stream>>>(Wq, Wq_h, nW);
  f32_to_f16_kernel<<<(nW + 255) / 256, 256, 0, stream>>>(Wk, Wk_h, nW);
  f32_to_f16_kernel<<<(nW + 255) / 256, 256, 0, stream>>>(Wv, Wv_h, nW);
  f32_to_f16_kernel<<<(nW + 255) / 256, 256, 0, stream>>>(Wo, Wo_h, nW);

  dim3 gg(DM / 64, (BSZ * SLN) / 128);
  gemm_wmma<true, true><<<gg, 128, 0, stream>>>(q, Wq_h, bq, qh, BSZ * SLN, DM, DM);
  gemm_wmma<true, true><<<gg, 128, 0, stream>>>(k, Wk_h, bk, khb, BSZ * SLN, DM, DM);
  gemm_wmma<true, true><<<gg, 128, 0, stream>>>(v, Wv_h, bv, vhb, BSZ * SLN, DM, DM);

  flash_attn<<<dim3(SLN / 64, BHN), 128, 0, stream>>>(qh, khb, vhb, mask, cat);

  gemm_wmma<false, false><<<gg, 128, 0, stream>>>(cat, Wo_h, bo, out, BSZ * SLN, DM, DM);
}
